// GraphTransformer_69114613730718
// MI455X (gfx1250) — matrix operand order repaired and checked by reading, not compiled, not run on hardware
//
#include <hip/hip_runtime.h>

typedef __bf16 bf16_t;
typedef __attribute__((ext_vector_type(8)))  __bf16 v8bf;
typedef __attribute__((ext_vector_type(16))) __bf16 v16bf;
typedef __attribute__((ext_vector_type(8)))  float  v8f;

// Problem constants (from reference)
#define NN   50000
#define NE   800000
#define CIN  128
#define HID  32
#define HEADS 8
#define D1   256      // HID*HEADS
#define NC1  1024     // 4*D1 fused Q|K|V|S
#define COUT 40
#define NC2  160      // 4*COUT fused

// ---------------- helpers ----------------
__device__ __forceinline__ bf16_t to_bf16(float f) {
    unsigned u = __float_as_uint(f);
    unsigned r = (u + 0x7FFFu + ((u >> 16) & 1u)) >> 16;
    unsigned short h = (unsigned short)r;
    return __builtin_bit_cast(bf16_t, h);
}
// order-preserving float<->uint map for atomicMax-based segment max
__device__ __forceinline__ unsigned f2mono(float f) {
    unsigned u = __float_as_uint(f);
    return (u & 0x80000000u) ? ~u : (u | 0x80000000u);
}
__device__ __forceinline__ float mono2f(unsigned u) {
    return __uint_as_float((u & 0x80000000u) ? (u ^ 0x80000000u) : ~u);
}
#define MONO_NEG_INF 0x007FFFFFu  // f2mono(-inf)

// ---------------- elementwise / packing ----------------
__global__ void k_f32_to_bf16(const float* __restrict__ src, bf16_t* __restrict__ dst, int n) {
    int i = blockIdx.x * blockDim.x + threadIdx.x;
    if (i < n) dst[i] = to_bf16(src[i]);
}

__global__ void k_fill_u32(unsigned* __restrict__ p, unsigned v, int n) {
    int i = blockIdx.x * blockDim.x + threadIdx.x;
    if (i < n) p[i] = v;
}

// Pack 4 row-major [K, subCols] f32 weight matrices into one col-major bf16
// buffer Wt[col][k] with col = m*subCols + c, plus fused bias vector.
__global__ void k_pack_w4(const float* __restrict__ w0, const float* __restrict__ w1,
                          const float* __restrict__ w2, const float* __restrict__ w3,
                          const float* __restrict__ b0, const float* __restrict__ b1,
                          const float* __restrict__ b2, const float* __restrict__ b3,
                          bf16_t* __restrict__ wt, float* __restrict__ bcat,
                          int K, int subCols) {
    int idx = blockIdx.x * blockDim.x + threadIdx.x;
    int total = 4 * subCols * K;
    if (idx >= total) return;
    int col = idx / K, k = idx - col * K;
    int m = col / subCols, c = col - m * subCols;
    const float* w = (m == 0) ? w0 : (m == 1) ? w1 : (m == 2) ? w2 : w3;
    wt[idx] = to_bf16(w[(size_t)k * subCols + c]);
    if (k == 0) {
        const float* b = (m == 0) ? b0 : (m == 1) ? b1 : (m == 2) ? b2 : b3;
        bcat[col] = b[c];
    }
}

// ---------------- bf16 WMMA GEMM: C[M,Ncols] = A[M,K] @ W + bias ----------------
// A: row-major bf16 [M,K];  Wt: col-major bf16 [Ncols][K];  C: f32 row-major.
// One wave computes one 16x16 tile; fragment layouts per CDNA5 ISA 7.12.2.
__global__ void k_gemm_wmma(const bf16_t* __restrict__ A, const bf16_t* __restrict__ Wt,
                            const float* __restrict__ bias, float* __restrict__ C,
                            int M, int K, int Ncols) {
    const int WPB  = blockDim.x >> 5;
    int lane  = threadIdx.x & 31;
    int wave  = threadIdx.x >> 5;
    int row0  = blockIdx.x * 16;
    int colTile = blockIdx.y * WPB + wave;
    if (colTile * 16 >= Ncols) return;       // uniform per wave -> EXEC stays all-ones
    int col0 = colTile * 16;
    int half = lane >> 4;
    int l15  = lane & 15;

    const bf16_t* aRow = A  + (size_t)(row0 + l15) * K;
    const bf16_t* bCol = Wt + (size_t)(col0 + l15) * K;

    v8f acc = {};
    for (int k0 = 0; k0 < K; k0 += 32) {
        // A 16x32: lane half selects K blocks {0..7,16..23} / {8..15,24..31}
        v8bf alo = *(const v8bf*)(aRow + k0 + 8 * half);
        v8bf ahi = *(const v8bf*)(aRow + k0 + 16 + 8 * half);
        v16bf a;
#pragma unroll
        for (int i = 0; i < 8; ++i) { a[i] = alo[i]; a[i + 8] = ahi[i]; }
        // B 32x16: lane half selects K blocks {0..7,16..23} / {8..15,24..31}, matching A
        v8bf blo = *(const v8bf*)(bCol + k0 + 8 * half);
        v8bf bhi = *(const v8bf*)(bCol + k0 + 16 + 8 * half);
        v16bf b;
#pragma unroll
        for (int i = 0; i < 8; ++i) { b[i] = blo[i]; b[i + 8] = bhi[i]; }
        acc = __builtin_amdgcn_wmma_f32_16x16x32_bf16(false, a, false, b,
                                                      (short)0, acc, false, false);
    }
    float bv = bias[col0 + l15];
#pragma unroll
    for (int v = 0; v < 8; ++v)
        C[(size_t)(row0 + v + 8 * half) * Ncols + col0 + l15] = acc[v] + bv;
}

// ---------------- edge attention ----------------
// alpha[e,h] = scale * dot(q[dst], k[src]); segment max into amax (mono uint)
__global__ void k_edge_alpha(const float* __restrict__ qkvs, int stride, int qoff, int koff,
                             const int* __restrict__ ei, int E, int H, int Cc, float scale,
                             float* __restrict__ alpha, unsigned* __restrict__ amax) {
    int idx = blockIdx.x * blockDim.x + threadIdx.x;
    if (idx >= E * H) return;
    int e = idx / H, h = idx - e * H;
    int s = ei[e], d = ei[E + e];
    const float4* q = (const float4*)(qkvs + (size_t)d * stride + qoff + h * Cc);
    const float4* k = (const float4*)(qkvs + (size_t)s * stride + koff + h * Cc);
    float acc = 0.f;
    int n4 = Cc >> 2;
    for (int i = 0; i < n4; ++i) {
        float4 a = q[i], b = k[i];
        acc += a.x * b.x + a.y * b.y + a.z * b.z + a.w * b.w;
    }
    acc *= scale;
    alpha[idx] = acc;
    atomicMax(&amax[(size_t)d * H + h], f2mono(acc));
}

__global__ void k_edge_expsum(float* __restrict__ alpha, const unsigned* __restrict__ amax,
                              float* __restrict__ asum, const int* __restrict__ ei,
                              int E, int H) {
    int idx = blockIdx.x * blockDim.x + threadIdx.x;
    if (idx >= E * H) return;
    int e = idx / H, h = idx - e * H;
    int d = ei[E + e];
    float a = __expf(alpha[idx] - mono2f(amax[(size_t)d * H + h]));
    alpha[idx] = a;
    atomicAdd(&asum[(size_t)d * H + h], a);
}

__global__ void k_edge_scatter(const float* __restrict__ alpha, const float* __restrict__ asum,
                               const float* __restrict__ qkvs, int stride, int voff,
                               float* __restrict__ aggr, const int* __restrict__ ei,
                               int E, int H, int Cc) {
    int idx = blockIdx.x * blockDim.x + threadIdx.x;
    if (idx >= E * H) return;
    int e = idx / H, h = idx - e * H;
    int s = ei[e], d = ei[E + e];
    float w = alpha[idx] / asum[(size_t)d * H + h];
    const float4* v = (const float4*)(qkvs + (size_t)s * stride + voff + h * Cc);
    float* o = aggr + (size_t)d * H * Cc + h * Cc;
    int n4 = Cc >> 2;
    for (int i = 0; i < n4; ++i) {
        float4 vv = v[i];
        atomicAdd(o + 4 * i + 0, vv.x * w);
        atomicAdd(o + 4 * i + 1, vv.y * w);
        atomicAdd(o + 4 * i + 2, vv.z * w);
        atomicAdd(o + 4 * i + 3, vv.w * w);
    }
}

// ---------------- beta gate (+optional ELU) ----------------
// beta = sigmoid(out.bw0 + skip.bw1 + (out-skip).bw2); res = beta*skip+(1-beta)*out
__global__ void k_gate(const float* __restrict__ aggr, const float* __restrict__ qkvs,
                       int stride, int soff, const float* __restrict__ bw, int D,
                       int n, int applyElu,
                       bf16_t* __restrict__ outB, float* __restrict__ outF) {
    int i = blockIdx.x * blockDim.x + threadIdx.x;
    if (i >= n) return;
    const float* o  = aggr + (size_t)i * D;
    const float* sk = qkvs + (size_t)i * stride + soff;
    float acc = 0.f;
    for (int c = 0; c < D; ++c) {
        float ov = o[c], sv = sk[c];
        acc += ov * bw[c] + sv * bw[D + c] + (ov - sv) * bw[2 * D + c];
    }
    float beta = 1.f / (1.f + __expf(-acc));
    for (int c = 0; c < D; ++c) {
        float ov = o[c], sv = sk[c];
        float val = beta * sv + (1.f - beta) * ov;
        if (applyElu) val = (val > 0.f) ? val : (__expf(val) - 1.f);
        if (outB) outB[(size_t)i * D + c] = to_bf16(val);
        else      outF[(size_t)i * D + c] = val;
    }
}

// ---------------- host launch ----------------
extern "C" void kernel_launch(void* const* d_in, const int* in_sizes, int n_in,
                              void* d_out, int out_size, void* d_ws, size_t ws_size,
                              hipStream_t stream) {
    const float* x   = (const float*)d_in[0];
    const int*   ei  = (const int*)d_in[1];
    const float* q1w = (const float*)d_in[2];  const float* q1b = (const float*)d_in[3];
    const float* k1w = (const float*)d_in[4];  const float* k1b = (const float*)d_in[5];
    const float* v1w = (const float*)d_in[6];  const float* v1b = (const float*)d_in[7];
    const float* s1w = (const float*)d_in[8];  const float* s1b = (const float*)d_in[9];
    const float* b1w = (const float*)d_in[10];
    const float* q2w = (const float*)d_in[11]; const float* q2b = (const float*)d_in[12];
    const float* k2w = (const float*)d_in[13]; const float* k2b = (const float*)d_in[14];
    const float* v2w = (const float*)d_in[15]; const float* v2b = (const float*)d_in[16];
    const float* s2w = (const float*)d_in[17]; const float* s2b = (const float*)d_in[18];
    const float* b2w = (const float*)d_in[19];
    float* outp = (float*)d_out;

    // bump allocator over workspace
    char* ws = (char*)d_ws;
    size_t off = 0;
    auto alloc = [&](size_t bytes) -> void* {
        void* p = ws + off;
        off = (off + bytes + 255) & ~(size_t)255;
        return p;
    };
    bf16_t*  xb    = (bf16_t*)alloc((size_t)NN * CIN * 2);
    bf16_t*  wt1   = (bf16_t*)alloc((size_t)NC1 * CIN * 2);
    float*   bcat1 = (float*) alloc((size_t)NC1 * 4);
    float*   qkvs1 = (float*) alloc((size_t)NN * NC1 * 4);
    float*   alpha1= (float*) alloc((size_t)NE * HEADS * 4);
    unsigned* amax1= (unsigned*)alloc((size_t)NN * HEADS * 4);
    float*   asum1 = (float*) alloc((size_t)NN * HEADS * 4);
    float*   aggr1 = (float*) alloc((size_t)NN * D1 * 4);
    bf16_t*  hb    = (bf16_t*)alloc((size_t)NN * D1 * 2);
    bf16_t*  wt2   = (bf16_t*)alloc((size_t)NC2 * D1 * 2);
    float*   bcat2 = (float*) alloc((size_t)NC2 * 4);
    float*   qkvs2 = (float*) alloc((size_t)NN * NC2 * 4);
    float*   alpha2= (float*) alloc((size_t)NE * 4);
    unsigned* amax2= (unsigned*)alloc((size_t)NN * 4);
    float*   asum2 = (float*) alloc((size_t)NN * 4);
    float*   aggr2 = (float*) alloc((size_t)NN * COUT * 4);

    const int T = 256;
    auto blocks = [](long long n, int t) { return (int)((n + t - 1) / t); };

    // ---- layer 1 ----
    k_f32_to_bf16<<<blocks((long long)NN * CIN, T), T, 0, stream>>>(x, xb, NN * CIN);
    k_pack_w4<<<blocks((long long)NC1 * CIN, T), T, 0, stream>>>(
        q1w, k1w, v1w, s1w, q1b, k1b, v1b, s1b, wt1, bcat1, CIN, D1);
    k_gemm_wmma<<<dim3(NN / 16, NC1 / 16 / 4), 128, 0, stream>>>(
        xb, wt1, bcat1, qkvs1, NN, CIN, NC1);

    k_fill_u32<<<blocks((long long)NN * HEADS, T), T, 0, stream>>>(amax1, MONO_NEG_INF, NN * HEADS);
    k_fill_u32<<<blocks((long long)NN * HEADS, T), T, 0, stream>>>((unsigned*)asum1, 0u, NN * HEADS);
    k_fill_u32<<<blocks((long long)NN * D1, T), T, 0, stream>>>((unsigned*)aggr1, 0u, NN * D1);

    k_edge_alpha<<<blocks((long long)NE * HEADS, T), T, 0, stream>>>(
        qkvs1, NC1, 0, D1, ei, NE, HEADS, HID, 0.17677669529663687f, alpha1, amax1);
    k_edge_expsum<<<blocks((long long)NE * HEADS, T), T, 0, stream>>>(
        alpha1, amax1, asum1, ei, NE, HEADS);
    k_edge_scatter<<<blocks((long long)NE * HEADS, T), T, 0, stream>>>(
        alpha1, asum1, qkvs1, NC1, 2 * D1, aggr1, ei, NE, HEADS, HID);
    k_gate<<<blocks(NN, T), T, 0, stream>>>(
        aggr1, qkvs1, NC1, 3 * D1, b1w, D1, NN, 1, hb, nullptr);

    // ---- layer 2 ----
    k_pack_w4<<<blocks((long long)NC2 * D1, T), T, 0, stream>>>(
        q2w, k2w, v2w, s2w, q2b, k2b, v2b, s2b, wt2, bcat2, D1, COUT);
    k_gemm_wmma<<<dim3(NN / 16, (NC2 / 16 + 3) / 4), 128, 0, stream>>>(
        hb, wt2, bcat2, qkvs2, NN, D1, NC2);

    k_fill_u32<<<blocks(NN, T), T, 0, stream>>>(amax2, MONO_NEG_INF, NN);
    k_fill_u32<<<blocks(NN, T), T, 0, stream>>>((unsigned*)asum2, 0u, NN);
    k_fill_u32<<<blocks((long long)NN * COUT, T), T, 0, stream>>>((unsigned*)aggr2, 0u, NN * COUT);

    k_edge_alpha<<<blocks(NE, T), T, 0, stream>>>(
        qkvs2, NC2, 0, COUT, ei, NE, 1, COUT, 0.15811388300841897f, alpha2, amax2);
    k_edge_expsum<<<blocks(NE, T), T, 0, stream>>>(alpha2, amax2, asum2, ei, NE, 1);
    k_edge_scatter<<<blocks(NE, T), T, 0, stream>>>(
        alpha2, asum2, qkvs2, NC2, 2 * COUT, aggr2, ei, NE, 1, COUT);
    k_gate<<<blocks(NN, T), T, 0, stream>>>(
        aggr2, qkvs2, NC2, 3 * COUT, b2w, COUT, NN, 0, nullptr, outp);
}
